// DGCNNBlock_38800734552598
// MI455X (gfx1250) — matrix operand order repaired and checked by reading, not compiled, not run on hardware
//
#include <hip/hip_runtime.h>
#include <hip/hip_bf16.h>

typedef __attribute__((ext_vector_type(16))) _Float16 v16h;
typedef __attribute__((ext_vector_type(8)))  _Float16 v8h;
typedef __attribute__((ext_vector_type(8)))  float    v8f;
typedef __attribute__((ext_vector_type(4)))  float    f32x4;

union AFrag { v16h v; v8h h[2]; };

// convert 8 consecutive f32 -> 8 f16
__device__ __forceinline__ v8h cvt8(const float* __restrict__ p) {
    f32x4 a = *(const f32x4*)p;
    f32x4 b = *(const f32x4*)(p + 4);
    v8h r;
    r[0] = (_Float16)a[0]; r[1] = (_Float16)a[1];
    r[2] = (_Float16)a[2]; r[3] = (_Float16)a[3];
    r[4] = (_Float16)b[0]; r[5] = (_Float16)b[1];
    r[6] = (_Float16)b[2]; r[7] = (_Float16)b[3];
    return r;
}

// convert 8 consecutive (pj[k] - pi[k]) -> 8 f16
__device__ __forceinline__ v8h cvt8diff(const float* __restrict__ pj,
                                        const float* __restrict__ pi) {
    f32x4 aj = *(const f32x4*)pj;       f32x4 bj = *(const f32x4*)(pj + 4);
    f32x4 ai = *(const f32x4*)pi;       f32x4 bi = *(const f32x4*)(pi + 4);
    v8h r;
    r[0] = (_Float16)(aj[0] - ai[0]); r[1] = (_Float16)(aj[1] - ai[1]);
    r[2] = (_Float16)(aj[2] - ai[2]); r[3] = (_Float16)(aj[3] - ai[3]);
    r[4] = (_Float16)(bj[0] - bi[0]); r[5] = (_Float16)(bj[1] - bi[1]);
    r[6] = (_Float16)(bj[2] - bi[2]); r[7] = (_Float16)(bj[3] - bi[3]);
    return r;
}

__global__ void init_neg_inf_kernel(float* __restrict__ out, int n) {
    int t = blockIdx.x * blockDim.x + threadIdx.x;
    if (t < n) out[t] = -__builtin_inff();
}

__global__ void finalize_kernel(float* __restrict__ out, int n) {
    int t = blockIdx.x * blockDim.x + threadIdx.x;
    if (t < n) {
        float v = out[t];
        if (v == -__builtin_inff()) out[t] = 0.0f;
    }
}

// One wave = one 16-edge tile. Block = 8 waves = 128 edges per iteration.
__global__ __launch_bounds__(256)
void edgeconv_wmma_kernel(const float* __restrict__ x,
                          const float* __restrict__ W1,   // [64][128]
                          const float* __restrict__ b1,   // [64]
                          const float* __restrict__ W2,   // [64][64]
                          const float* __restrict__ b2,   // [64]
                          const int*   __restrict__ eidx, // [2][E] (src row 0, dst row 1)
                          float*       __restrict__ out,  // [N][64]
                          int nEdges) {
    __shared__ __align__(16) _Float16 sW1[64][128];     // 16 KB, f16 copy of W1
    __shared__ __align__(16) _Float16 sW2[64][64];      //  8 KB, f16 copy of W2
    __shared__ __align__(16) _Float16 sH[8][16][64];    // 16 KB, per-wave h staging

    const int tid = threadIdx.x;

    // Stage weights to LDS as f16 (once per block).
    for (int t = tid; t < 64 * 128; t += 256)
        sW1[t >> 7][t & 127] = (_Float16)W1[t];
    for (int t = tid; t < 64 * 64; t += 256)
        sW2[t >> 6][t & 63] = (_Float16)W2[t];
    __syncthreads();

    const int wave = tid >> 5;
    const int lane = tid & 31;
    const int lhi  = lane >> 4;        // 0 | 1 (lane half)
    const int l16  = lane & 15;
    const int klo  = lhi << 3;         // 0 | 8   (A-fragment K sub-offset)

    const float b1v[4] = { b1[l16], b1[16 + l16], b1[32 + l16], b1[48 + l16] };
    const float b2v[4] = { b2[l16], b2[16 + l16], b2[32 + l16], b2[48 + l16] };

    const int tiles = (nEdges + 127) >> 7;
    for (int tb = blockIdx.x; tb < tiles; tb += gridDim.x) {
        const int e0 = (tb << 7) + (wave << 4);

        // --- gather the A-row this lane feeds (row M = l16) ---
        int eA = e0 + l16;
        if (eA >= nEdges) eA = nEdges - 1;
        const long iA = (long)eidx[nEdges + eA];   // dst (i)
        const long jA = (long)eidx[eA];            // src (j)
        const float* __restrict__ xi = x + iA * 64;
        const float* __restrict__ xj = x + jA * 64;

        // ---------------- GEMM1: h = relu([xi | xj-xi] @ W1^T + b1) ----------------
        v8f acc[4] = {};
        #pragma unroll
        for (int kb = 0; kb < 2; ++kb) {           // K = 0..63 : xi part
            const int base = kb * 32;
            AFrag a;
            a.h[0] = cvt8(xi + base + klo);
            a.h[1] = cvt8(xi + base + 16 + klo);
            #pragma unroll
            for (int nb = 0; nb < 4; ++nb) {
                AFrag bf;
                const _Float16* bp = &sW1[nb * 16 + l16][base + (lhi << 3)];
                bf.h[0] = *(const v8h*)bp;
                bf.h[1] = *(const v8h*)(bp + 16);
                acc[nb] = __builtin_amdgcn_wmma_f32_16x16x32_f16(
                    false, a.v, false, bf.v, (short)0, acc[nb], false, false);
            }
        }
        #pragma unroll
        for (int kb = 0; kb < 2; ++kb) {           // K = 64..127 : (xj - xi) part
            const int base = kb * 32;
            AFrag a;
            a.h[0] = cvt8diff(xj + base + klo,      xi + base + klo);
            a.h[1] = cvt8diff(xj + base + 16 + klo, xi + base + 16 + klo);
            #pragma unroll
            for (int nb = 0; nb < 4; ++nb) {
                AFrag bf;
                const _Float16* bp = &sW1[nb * 16 + l16][64 + base + (lhi << 3)];
                bf.h[0] = *(const v8h*)bp;
                bf.h[1] = *(const v8h*)(bp + 16);
                acc[nb] = __builtin_amdgcn_wmma_f32_16x16x32_f16(
                    false, a.v, false, bf.v, (short)0, acc[nb], false, false);
            }
        }

        // bias + relu, stage h into per-wave LDS in row-major [16][64] f16
        #pragma unroll
        for (int nb = 0; nb < 4; ++nb) {
            #pragma unroll
            for (int g = 0; g < 8; ++g) {
                float h = acc[nb][g] + b1v[nb];
                h = h > 0.0f ? h : 0.0f;
                sH[wave][g + (lhi << 3)][nb * 16 + l16] = (_Float16)h;
            }
        }
        // all lanes of this wave executed the stores above; drain DS pipe
        asm volatile("s_wait_dscnt 0x0" ::: "memory");

        // ---------------- GEMM2: e = h @ W2^T + b2 ----------------
        v8f acc2[4] = {};
        #pragma unroll
        for (int kb = 0; kb < 2; ++kb) {
            AFrag a;
            const _Float16* ap = &sH[wave][l16][kb * 32 + klo];
            a.h[0] = *(const v8h*)ap;
            a.h[1] = *(const v8h*)(ap + 16);
            #pragma unroll
            for (int nb = 0; nb < 4; ++nb) {
                AFrag bf;
                const _Float16* bp = &sW2[nb * 16 + l16][kb * 32 + (lhi << 3)];
                bf.h[0] = *(const v8h*)bp;
                bf.h[1] = *(const v8h*)(bp + 16);
                acc2[nb] = __builtin_amdgcn_wmma_f32_16x16x32_f16(
                    false, a.v, false, bf.v, (short)0, acc2[nb], false, false);
            }
        }

        // ---------------- scatter: segment-max via native f32 atomic max ----------------
        #pragma unroll
        for (int g = 0; g < 8; ++g) {
            const int er    = e0 + g + (lhi << 3);      // edge row of accumulator g
            const bool valid = er < nEdges;
            const int dsti  = eidx[nEdges + (valid ? er : 0)];
            float* rowp = out + (long)dsti * 64 + l16;
            #pragma unroll
            for (int nb = 0; nb < 4; ++nb) {
                float v = acc2[nb][g] + b2v[nb];
                if (valid) {
                    float* addr = rowp + nb * 16;
                    asm volatile("global_atomic_max_num_f32 %0, %1, off scope:SCOPE_DEV"
                                 :: "v"(addr), "v"(v) : "memory");
                }
            }
        }
    }
}

extern "C" void kernel_launch(void* const* d_in, const int* in_sizes, int n_in,
                              void* d_out, int out_size, void* d_ws, size_t ws_size,
                              hipStream_t stream) {
    const float* x   = (const float*)d_in[0];
    const float* W1  = (const float*)d_in[1];
    const float* b1  = (const float*)d_in[2];
    const float* W2  = (const float*)d_in[3];
    const float* b2  = (const float*)d_in[4];
    const int*  eidx = (const int*)d_in[5];
    float* out = (float*)d_out;

    const int nEdges = in_sizes[5] / 2;

    // 1) init output to -inf (so pure-max aggregation is correct for negative maxima)
    init_neg_inf_kernel<<<(out_size + 255) / 256, 256, 0, stream>>>(out, out_size);

    // 2) fused gather + MLP(WMMA) + atomic-max scatter
    const int tiles = (nEdges + 127) / 128;
    int grid = tiles < 1024 ? tiles : 1024;
    if (grid < 1) grid = 1;
    edgeconv_wmma_kernel<<<grid, 256, 0, stream>>>(x, W1, b1, W2, b2, eidx, out, nEdges);

    // 3) nodes with no incoming edges: -inf -> 0 (matches reference where())
    finalize_kernel<<<(out_size + 255) / 256, 256, 0, stream>>>(out, out_size);
}
